// DiscSeqRNNEncoder_17119739641881
// MI455X (gfx1250) — matrix operand order repaired and checked by reading, not compiled, not run on hardware
//
#include <hip/hip_runtime.h>
#include <hip/hip_bf16.h>

// ---------------------------------------------------------------------------
// Fused GRU encoder for MI455X (gfx1250, wave32, WMMA).
//   out[B,128] = GRU(embed[x], W_ih, W_hh, b_ih, b_hh) @ W_out.T + b_out
// One wave = 16 batch rows held as f32 WMMA accumulators for the whole
// sequence; per-step gate GEMM runs on v_wmma_f32_16x16x32_bf16 with weight
// B-fragments resident in LDS.
// ---------------------------------------------------------------------------

typedef __attribute__((ext_vector_type(16))) __bf16 bf16x16;
typedef __attribute__((ext_vector_type(8)))  __bf16 bf16x8;
typedef __attribute__((ext_vector_type(4)))  __bf16 bf16x4;
typedef __attribute__((ext_vector_type(8)))  float  f32x8;

#define B_TOT   8192
#define LSEQ    64
#define EMB     64
#define HID     128
#define OUTF    128

#define WAVES           4
#define THREADS         (WAVES * 32)
#define ROWS_PER_WAVE   16
#define ROWS_PER_BLOCK  (WAVES * ROWS_PER_WAVE)
#define NBLOCKS         (B_TOT / ROWS_PER_BLOCK)

#define KT_C        6            // combined K = [E(64) | H(128)] = 192 -> 6 tiles of 32
#define NT_C        24           // 3H = 384 -> 24 tiles of 16
#define KT_O        4            // H = 128  -> 4 tiles of 32
#define NT_O        8            // OUT = 128 -> 8 tiles of 16
#define FRAG_ELEMS  512          // 32 lanes * 16 bf16 per fragment tile

// LDS map (in bf16 elements)
#define WC_OFF      0
#define WC_ELEMS    (KT_C * NT_C * FRAG_ELEMS)   // 73728  (144 KB)
#define WO_OFF      (WC_OFF + WC_ELEMS)
#define WO_ELEMS    (KT_O * NT_O * FRAG_ELEMS)   // 16384  (32 KB)
#define HST_OFF     (WO_OFF + WO_ELEMS)
#define HSTRIDE     136                          // 128 cols + 8 pad (kills bank conflicts)
#define HST_ELEMS   (WAVES * 16 * HSTRIDE)       // 8704   (17 KB)
#define SMEM_ELEMS  (HST_OFF + HST_ELEMS)
#define SMEM_BYTES  (SMEM_ELEMS * 2)             // 197632 B < 320 KB / WGP

// --------------------------- small device helpers ---------------------------

__device__ __forceinline__ f32x8 splat8(float v) {
  f32x8 r;
#pragma unroll
  for (int i = 0; i < 8; ++i) r[i] = v;
  return r;
}

__device__ __forceinline__ f32x8 wmma_bf16(bf16x16 a, bf16x16 b, f32x8 c) {
  // v_wmma_f32_16x16x32_bf16: D = A(16x32) * B(32x16) + C
  return __builtin_amdgcn_wmma_f32_16x16x32_bf16(
      /*neg_a=*/false, a, /*neg_b=*/false, b,
      /*c_mod=*/(short)0, c, /*reuse_a=*/false, /*reuse_b=*/false);
}

// B-fragment: stored in LDS as [4 qwords][32 lanes][4 bf16] per tile so every
// lane issues 4 conflict-free ds_load_b64 (bank = 2*lane, 2*lane+1).
__device__ __forceinline__ bf16x16 load_bfrag(const __bf16* tbase, int lane) {
  const bf16x4* p = (const bf16x4*)tbase;
  bf16x4 q0 = p[0 * 32 + lane];
  bf16x4 q1 = p[1 * 32 + lane];
  bf16x4 q2 = p[2 * 32 + lane];
  bf16x4 q3 = p[3 * 32 + lane];
  bf16x16 r;
#pragma unroll
  for (int i = 0; i < 4; ++i) {
    r[i] = q0[i]; r[4 + i] = q1[i]; r[8 + i] = q2[i]; r[12 + i] = q3[i];
  }
  return r;
}

// A-fragment of h from the per-wave LDS stage buffer (row-major, stride 136).
// A 16-bit 16x32 layout: lane holds row (lane&15); element e -> K = e+(e&8)+8*hi,
// i.e. two contiguous 8-element (16 B) chunks at K = hi*8 and K = 16+hi*8.
__device__ __forceinline__ bf16x16 load_afrag(const __bf16* rowp, int kt, int hi) {
  bf16x8 c0 = *(const bf16x8*)(rowp + kt * 32 + hi * 8);
  bf16x8 c1 = *(const bf16x8*)(rowp + kt * 32 + 16 + hi * 8);
  bf16x16 r;
#pragma unroll
  for (int i = 0; i < 8; ++i) { r[i] = c0[i]; r[8 + i] = c1[i]; }
  return r;
}

__device__ __forceinline__ float fast_sigmoid(float x) {
  x = fminf(fmaxf(x, -30.f), 30.f);
  return __builtin_amdgcn_rcpf(1.f + __expf(-x));   // v_exp + v_rcp (trans, co-exec w/ XDL)
}
__device__ __forceinline__ float fast_tanh(float x) {
  x = fminf(fmaxf(x, -15.f), 15.f);
  float e = __expf(-2.f * x);
  return (1.f - e) * __builtin_amdgcn_rcpf(1.f + e);
}

// ---------------------------------- kernel ----------------------------------

extern "C" __global__ __launch_bounds__(THREADS, 1)
void gru_encode_wmma_kernel(const int* __restrict__ x,
                            const float* __restrict__ embed,
                            const float* __restrict__ W_ih,
                            const float* __restrict__ W_hh,
                            const float* __restrict__ b_ih,
                            const float* __restrict__ b_hh,
                            const float* __restrict__ W_out,
                            const float* __restrict__ b_out,
                            float* __restrict__ out) {
  extern __shared__ __bf16 smem[];
  const int tid  = threadIdx.x;
  const int lane = tid & 31;
  const int wave = tid >> 5;
  const int col  = lane & 15;   // N index inside a 16-wide tile / row for A frags
  const int hi   = lane >> 4;   // lane-half selector

  // ---- Pack weight B-fragments into LDS (once per block). -----------------
  // Combined B matrix Bc[k][n], k in [0,192): k<64 -> W_ih[n][k], else W_hh[n][k-64].
  // B 16-bit 32x16 layout: lane holds col (l&15); element e -> K = (e&7) + 8*hi + 2*(e&8).
  for (int f = tid; f < WC_ELEMS; f += THREADS) {
    int tile = f >> 9, rem = f & 511;
    int l = (rem >> 2) & 31, q = rem >> 7, m = rem & 3;
    int e  = q * 4 + m;
    int kt = tile / NT_C, nt = tile % NT_C;
    int hh = l >> 4;
    int n  = nt * 16 + (l & 15);
    int kk = kt * 32 + (e & 7) + hh * 8 + (e & 8) * 2;
    float v = (kk < EMB) ? W_ih[n * EMB + kk] : W_hh[n * HID + (kk - EMB)];
    smem[WC_OFF + f] = (__bf16)v;
  }
  // Output-projection fragments: Bo[k][n] = W_out[n][k].
  for (int f = tid; f < WO_ELEMS; f += THREADS) {
    int tile = f >> 9, rem = f & 511;
    int l = (rem >> 2) & 31, q = rem >> 7, m = rem & 3;
    int e  = q * 4 + m;
    int kt = tile / NT_O, nt = tile % NT_O;
    int hh = l >> 4;
    int n  = nt * 16 + (l & 15);
    int kk = kt * 32 + (e & 7) + hh * 8 + (e & 8) * 2;
    smem[WO_OFF + f] = (__bf16)W_out[n * HID + kk];
  }
  __syncthreads();

  const __bf16* wc  = smem + WC_OFF;
  const __bf16* wo  = smem + WO_OFF;
  __bf16*       hst = smem + HST_OFF + wave * 16 * HSTRIDE;  // wave-private

  const int rowBase = blockIdx.x * ROWS_PER_BLOCK + wave * ROWS_PER_WAVE;
  const int gRow    = rowBase + col;   // batch row this lane gathers for

  // Per-lane gate biases (PyTorch GRU: r/z biases merge; n-gate stays split
  // because n = tanh(xn + r * hn) with hn carrying b_hh_n).
  float bR[8], bZ[8], bXN[8], bHN[8];
#pragma unroll
  for (int j = 0; j < 8; ++j) {
    int c  = j * 16 + col;
    bR[j]  = b_ih[c]       + b_hh[c];
    bZ[j]  = b_ih[128 + c] + b_hh[128 + c];
    bXN[j] = b_ih[256 + c];
    bHN[j] = b_hh[256 + c];
  }

  // Hidden state: 8 f32 C-fragments (element r of lane = h[M=r+8*hi][N=j*16+col]).
  f32x8 h[8];
#pragma unroll
  for (int j = 0; j < 8; ++j) h[j] = splat8(0.f);

  typedef __attribute__((ext_vector_type(8))) float f32v8g;

#pragma unroll 1
  for (int t = 0; t < LSEQ; ++t) {
    // -- Stage h (bf16) into wave-private LDS for A-fragment reads. ---------
#pragma unroll
    for (int j = 0; j < 8; ++j) {
#pragma unroll
      for (int r = 0; r < 8; ++r) {
        hst[(r + hi * 8) * HSTRIDE + j * 16 + col] = (__bf16)h[j][r];
      }
    }

    // -- Gather this step's embedding row into two A-fragments. -------------
    int idx  = x[gRow * LSEQ + t];
    int idx2 = x[gRow * LSEQ + ((t + 1 < LSEQ) ? (t + 1) : t)];
    __builtin_prefetch(embed + (size_t)idx2 * EMB, 0, 0);   // global_prefetch_b8

    const float* er = embed + (size_t)idx * EMB + hi * 8;
    f32v8g c0 = *(const f32v8g*)(er);        // K = hi*8   .. +7
    f32v8g c1 = *(const f32v8g*)(er + 16);   // K = 16+hi*8.. +7
    f32v8g c2 = *(const f32v8g*)(er + 32);   // K = 32+hi*8.. +7
    f32v8g c3 = *(const f32v8g*)(er + 48);   // K = 48+hi*8.. +7
    bf16x16 ae0, ae1;
#pragma unroll
    for (int i = 0; i < 8; ++i) {
      ae0[i]     = (__bf16)c0[i];
      ae0[8 + i] = (__bf16)c1[i];
      ae1[i]     = (__bf16)c2[i];
      ae1[8 + i] = (__bf16)c3[i];
    }

    // -- h A-fragments (DS ops are wave-ordered; no barrier needed). --------
    const __bf16* rowp = hst + col * HSTRIDE;
    bf16x16 ah0 = load_afrag(rowp, 0, hi);
    bf16x16 ah1 = load_afrag(rowp, 1, hi);
    bf16x16 ah2 = load_afrag(rowp, 2, hi);
    bf16x16 ah3 = load_afrag(rowp, 3, hi);

#define BFRAG(KT, NT) load_bfrag(wc + ((KT) * NT_C + (NT)) * FRAG_ELEMS, lane)

    // -- Gate GEMMs + fused elementwise GRU update, tile by tile. -----------
#pragma unroll
    for (int j = 0; j < 8; ++j) {
      f32x8 aR  = splat8(bR[j]);
      f32x8 aZ  = splat8(bZ[j]);
      f32x8 aXN = splat8(bXN[j]);
      f32x8 aHN = splat8(bHN[j]);

      aR = wmma_bf16(ae0, BFRAG(0, j), aR);
      aR = wmma_bf16(ae1, BFRAG(1, j), aR);
      aR = wmma_bf16(ah0, BFRAG(2, j), aR);
      aR = wmma_bf16(ah1, BFRAG(3, j), aR);
      aR = wmma_bf16(ah2, BFRAG(4, j), aR);
      aR = wmma_bf16(ah3, BFRAG(5, j), aR);

      aZ = wmma_bf16(ae0, BFRAG(0, 8 + j), aZ);
      aZ = wmma_bf16(ae1, BFRAG(1, 8 + j), aZ);
      aZ = wmma_bf16(ah0, BFRAG(2, 8 + j), aZ);
      aZ = wmma_bf16(ah1, BFRAG(3, 8 + j), aZ);
      aZ = wmma_bf16(ah2, BFRAG(4, 8 + j), aZ);
      aZ = wmma_bf16(ah3, BFRAG(5, 8 + j), aZ);

      aXN = wmma_bf16(ae0, BFRAG(0, 16 + j), aXN);
      aXN = wmma_bf16(ae1, BFRAG(1, 16 + j), aXN);

      aHN = wmma_bf16(ah0, BFRAG(2, 16 + j), aHN);
      aHN = wmma_bf16(ah1, BFRAG(3, 16 + j), aHN);
      aHN = wmma_bf16(ah2, BFRAG(4, 16 + j), aHN);
      aHN = wmma_bf16(ah3, BFRAG(5, 16 + j), aHN);

#pragma unroll
      for (int r = 0; r < 8; ++r) {
        float rg = fast_sigmoid(aR[r]);
        float zg = fast_sigmoid(aZ[r]);
        float ng = fast_tanh(aXN[r] + rg * aHN[r]);
        h[j][r]  = ng + zg * (h[j][r] - ng);   // (1-z)*n + z*h
      }
    }
  }

  // ---- Output projection: out = h @ W_out.T + b_out (same machinery). -----
#pragma unroll
  for (int j = 0; j < 8; ++j) {
#pragma unroll
    for (int r = 0; r < 8; ++r) {
      hst[(r + hi * 8) * HSTRIDE + j * 16 + col] = (__bf16)h[j][r];
    }
  }
  {
    const __bf16* rowp = hst + col * HSTRIDE;
    bf16x16 ah0 = load_afrag(rowp, 0, hi);
    bf16x16 ah1 = load_afrag(rowp, 1, hi);
    bf16x16 ah2 = load_afrag(rowp, 2, hi);
    bf16x16 ah3 = load_afrag(rowp, 3, hi);
#pragma unroll
    for (int nt = 0; nt < NT_O; ++nt) {
      f32x8 acc = splat8(b_out[nt * 16 + col]);
      acc = wmma_bf16(ah0, load_bfrag(wo + (0 * NT_O + nt) * FRAG_ELEMS, lane), acc);
      acc = wmma_bf16(ah1, load_bfrag(wo + (1 * NT_O + nt) * FRAG_ELEMS, lane), acc);
      acc = wmma_bf16(ah2, load_bfrag(wo + (2 * NT_O + nt) * FRAG_ELEMS, lane), acc);
      acc = wmma_bf16(ah3, load_bfrag(wo + (3 * NT_O + nt) * FRAG_ELEMS, lane), acc);
#pragma unroll
      for (int r = 0; r < 8; ++r) {
        out[(size_t)(rowBase + r + hi * 8) * OUTF + nt * 16 + col] = acc[r];
      }
    }
  }
#undef BFRAG
}

// --------------------------------- launcher ---------------------------------

extern "C" void kernel_launch(void* const* d_in, const int* in_sizes, int n_in,
                              void* d_out, int out_size, void* d_ws, size_t ws_size,
                              hipStream_t stream) {
  (void)in_sizes; (void)n_in; (void)out_size; (void)d_ws; (void)ws_size;
  const int*   x     = (const int*)d_in[0];
  const float* embed = (const float*)d_in[1];
  const float* W_ih  = (const float*)d_in[2];
  const float* W_hh  = (const float*)d_in[3];
  const float* b_ih  = (const float*)d_in[4];
  const float* b_hh  = (const float*)d_in[5];
  const float* W_out = (const float*)d_in[6];
  const float* b_out = (const float*)d_in[7];
  float* out = (float*)d_out;

  // Dynamic LDS > 64 KB: raise the limit (idempotent, capture-safe host call).
  (void)hipFuncSetAttribute((const void*)gru_encode_wmma_kernel,
                            hipFuncAttributeMaxDynamicSharedMemorySize, SMEM_BYTES);

  dim3 grid(NBLOCKS), block(THREADS);
  gru_encode_wmma_kernel<<<grid, block, SMEM_BYTES, stream>>>(
      x, embed, W_ih, W_hh, b_ih, b_hh, W_out, b_out, out);
}
